// GCM_83966610637414
// MI455X (gfx1250) — matrix operand order repaired and checked by reading, not compiled, not run on hardware
//
#include <hip/hip_runtime.h>
#include <hip/hip_bf16.h>

// ---------------- problem constants ----------------
#define BATCH 8
#define CDIM  512      // C
#define C4    128      // C/4
#define C2    256      // C/2
#define HWD   9216     // 96*96

typedef __attribute__((ext_vector_type(8)))  float  v8f;
typedef __attribute__((ext_vector_type(16))) __bf16 v16bf;
typedef __attribute__((ext_vector_type(8)))  __bf16 v8bf;

// float -> bf16, round-to-nearest-even (pure bit ops)
__device__ __forceinline__ __bf16 f2bf(float f) {
    union { float f; unsigned u; } in; in.f = f;
    unsigned u = in.u + 0x7FFFu + ((in.u >> 16) & 1u);
    unsigned short h = (unsigned short)(u >> 16);
    __bf16 r;
    __builtin_memcpy(&r, &h, 2);
    return r;
}

// ---------------- WMMA fragment loaders (16x16x32 bf16; contiguous k = 16h + s, same order for A and B) ----------------
// A 16x32 (MxK), source row-major [M][K], leading dim lda.
// Lane L<16 : row M=L,    K = k0..k0+15
// Lane L>=16: row M=L-16, K = k0+16..k0+31
__device__ __forceinline__ v16bf load_a_frag(const __bf16* A, int m0, int lda,
                                             int k0, int lane) {
    int r  = m0 + (lane & 15);
    int ko = k0 + ((lane >> 4) << 4);
    const __bf16* p = A + (long)r * lda + ko;
    v8bf lo = *(const v8bf*)(p);
    v8bf hi = *(const v8bf*)(p + 8);
    return __builtin_shufflevector(lo, hi, 0,1,2,3,4,5,6,7,8,9,10,11,12,13,14,15);
}

// B 32x16 (KxN), source is B-transposed row-major [N][K], leading dim ldb.
// Lane L<16 : col N=L,    K = k0..k0+15   (contiguous 32 B)
// Lane L>=16: col N=L-16, K = k0+16..k0+31
__device__ __forceinline__ v16bf load_b_frag(const __bf16* Bt, int n0, int ldb,
                                             int k0, int lane) {
    int r  = n0 + (lane & 15);
    int ko = k0 + ((lane >> 4) << 4);
    const __bf16* p = Bt + (long)r * ldb + ko;
    return *(const v16bf*)p;
}

__device__ __forceinline__ v8f wmma_bf16(v16bf a, v16bf b, v8f c) {
    return __builtin_amdgcn_wmma_f32_16x16x32_bf16(false, a, false, b,
                                                   (short)0, c, false, false);
}

// Register-blocked GEMM: one wave computes an (MT*16) x (NT*16) output block.
// Per K-step: MT + NT fragment loads feed MT*NT WMMAs.
template<int MT, int NT>
__device__ __forceinline__ void gemm_block(const __bf16* __restrict__ A, int m0, int lda,
                                           const __bf16* __restrict__ Bt, int n0, int ldb,
                                           int K, int lane, v8f (&acc)[MT][NT]) {
#pragma unroll
    for (int i = 0; i < MT; ++i)
#pragma unroll
        for (int j = 0; j < NT; ++j) acc[i][j] = v8f{};
    for (int k0 = 0; k0 < K; k0 += 32) {
        v16bf a[MT], b[NT];
#pragma unroll
        for (int i = 0; i < MT; ++i) a[i] = load_a_frag(A, m0 + 16 * i, lda, k0, lane);
#pragma unroll
        for (int j = 0; j < NT; ++j) b[j] = load_b_frag(Bt, n0 + 16 * j, ldb, k0, lane);
#pragma unroll
        for (int i = 0; i < MT; ++i)
#pragma unroll
            for (int j = 0; j < NT; ++j)
                acc[i][j] = wmma_bf16(a[i], b[j], acc[i][j]);
    }
}

// Single-tile GEMM (kept for the small Gram matrix, to preserve wave count)
__device__ __forceinline__ v8f gemm_tile(const __bf16* A, int m0, int lda,
                                         const __bf16* Bt, int n0, int ldb,
                                         int K, int lane) {
    v8f acc = {};
    for (int k0 = 0; k0 < K; k0 += 32) {
        v16bf a = load_a_frag(A, m0, lda, k0, lane);
        v16bf b = load_b_frag(Bt, n0, ldb, k0, lane);
        acc = wmma_bf16(a, b, acc);
    }
    return acc;
}

// D layout: VGPR i, lane L -> (M = mtile + i + 8*(L>=16), N = ntile + (L&15))

// ---------------- prep kernels ----------------
__global__ void k_cvt(const float* __restrict__ src, __bf16* __restrict__ dst, int n) {
    int i = blockIdx.x * blockDim.x + threadIdx.x;
    if (i < n) dst[i] = f2bf(src[i]);
}

// x [B][C][HW] fp32 -> xT [B][HW][C] bf16 (tiled transpose + convert)
__global__ void k_transpose(const float* __restrict__ x, __bf16* __restrict__ xT) {
    __shared__ __bf16 tile[32][33];
    int b  = blockIdx.z;
    int n0 = blockIdx.x * 32;
    int c0 = blockIdx.y * 32;
    const float* xb = x  + (long)b * CDIM * HWD;
    __bf16*     xTb = xT + (long)b * HWD * CDIM;
    int tx = threadIdx.x, ty = threadIdx.y;
    tile[ty][tx] = f2bf(xb[(long)(c0 + ty) * HWD + n0 + tx]);
    __syncthreads();
    xTb[(long)(n0 + ty) * CDIM + c0 + tx] = tile[tx][ty];
}

// ---------------- register-blocked GEMM kernels (256 thr = 8 waves, 64x32 / wave) ----------------

// conv1: x1[k][n] = W1 . x[:,n] + b1 ; also writes x1T[n][k] (packed)
__global__ void k_gemm_x1(const __bf16* __restrict__ w1b, const float* __restrict__ b1,
                          const __bf16* __restrict__ xT,
                          __bf16* __restrict__ x1, __bf16* __restrict__ x1T) {
    const int BM = C4 / 64, BN = HWD / 32;
    int wid  = (blockIdx.x * blockDim.x + threadIdx.x) >> 5;
    int lane = threadIdx.x & 31;
    int b  = wid / (BM * BN);
    int r  = wid % (BM * BN);
    int m0 = (r / BN) * 64, n0 = (r % BN) * 32;
    const __bf16* Bt = xT + (long)b * HWD * CDIM;
    v8f acc[4][2];
    gemm_block<4, 2>(w1b, m0, CDIM, Bt, n0, CDIM, CDIM, lane, acc);
    __bf16* x1b  = x1  + (long)b * C4 * HWD;
    __bf16* x1Tb = x1T + (long)b * HWD * C4;
#pragma unroll
    for (int i = 0; i < 4; ++i)
#pragma unroll
        for (int j = 0; j < 2; ++j) {
            int n     = n0 + 16 * j + (lane & 15);
            int mbase = m0 + 16 * i + ((lane >> 4) << 3);
            v8bf pack;
#pragma unroll
            for (int e = 0; e < 8; ++e) {
                __bf16 bv = f2bf(acc[i][j][e] + b1[mbase + e]);
                x1b[(long)(mbase + e) * HWD + n] = bv;
                pack[e] = bv;
            }
            *(v8bf*)(x1Tb + (long)n * C4 + mbase) = pack;
        }
}

// conv2: x2[m][n] = W2 . x[:,n] + b2
__global__ void k_gemm_x2(const __bf16* __restrict__ w2b, const float* __restrict__ b2,
                          const __bf16* __restrict__ xT, __bf16* __restrict__ x2) {
    const int BM = C2 / 64, BN = HWD / 32;
    int wid  = (blockIdx.x * blockDim.x + threadIdx.x) >> 5;
    int lane = threadIdx.x & 31;
    int b  = wid / (BM * BN);
    int r  = wid % (BM * BN);
    int m0 = (r / BN) * 64, n0 = (r % BN) * 32;
    const __bf16* Bt = xT + (long)b * HWD * CDIM;
    v8f acc[4][2];
    gemm_block<4, 2>(w2b, m0, CDIM, Bt, n0, CDIM, CDIM, lane, acc);
    __bf16* x2b = x2 + (long)b * C2 * HWD;
#pragma unroll
    for (int i = 0; i < 4; ++i)
#pragma unroll
        for (int j = 0; j < 2; ++j) {
            int n     = n0 + 16 * j + (lane & 15);
            int mbase = m0 + 16 * i + ((lane >> 4) << 3);
#pragma unroll
            for (int e = 0; e < 8; ++e)
                x2b[(long)(mbase + e) * HWD + n] = f2bf(acc[i][j][e] + b2[mbase + e]);
        }
}

// Gram: v[k][m] = sum_n x1[k][n] x2[m][n]  (K = 9216; 1 wave / 16x16 tile for parallelism)
__global__ void k_gemm_v(const __bf16* __restrict__ x1, const __bf16* __restrict__ x2,
                         float* __restrict__ v) {
    const int TM = C4 / 16, TN = C2 / 16;
    int wid  = (blockIdx.x * blockDim.x + threadIdx.x) >> 5;
    int lane = threadIdx.x & 31;
    int b  = wid / (TM * TN);
    int r  = wid % (TM * TN);
    int tm = r / TN, tn = r % TN;
    const __bf16* A  = x1 + (long)b * C4 * HWD;
    const __bf16* Bt = x2 + (long)b * C2 * HWD;
    v8f acc = gemm_tile(A, tm * 16, HWD, Bt, tn * 16, HWD, HWD, lane);
    int n     = tn * 16 + (lane & 15);
    int mbase = tm * 16 + ((lane >> 4) << 3);
    float* vb = v + (long)b * C4 * C2;
#pragma unroll
    for (int e = 0; e < 8; ++e)
        vb[(long)(mbase + e) * C2 + n] = acc[e];
}

// ---------------- small chain: softmax -> conv3+res+relu -> conv4 (1 block / batch) ----------------
__global__ void k_small(float* __restrict__ v,
                        const float* __restrict__ w3, const float* __restrict__ b3,
                        const float* __restrict__ w4, const float* __restrict__ b4,
                        float* __restrict__ nbuf, __bf16* __restrict__ z) {
    const int NV = C4 * C2;          // 32768
    int b   = blockIdx.x;
    int tid = threadIdx.x;
    float* vb = v + (long)b * NV;
    __shared__ float red[1024];

    float m = -3.0e38f;
    for (int i = tid; i < NV; i += 1024) m = fmaxf(m, vb[i]);
    red[tid] = m; __syncthreads();
    for (int s = 512; s > 0; s >>= 1) {
        if (tid < s) red[tid] = fmaxf(red[tid], red[tid + s]);
        __syncthreads();
    }
    float gmax = red[0]; __syncthreads();

    float sum = 0.f;
    for (int i = tid; i < NV; i += 1024) sum += __expf(vb[i] - gmax);
    red[tid] = sum; __syncthreads();
    for (int s = 512; s > 0; s >>= 1) {
        if (tid < s) red[tid] += red[tid + s];
        __syncthreads();
    }
    float inv = 1.0f / red[0]; __syncthreads();

    for (int i = tid; i < NV; i += 1024) vb[i] = __expf(vb[i] - gmax) * inv;
    __threadfence(); __syncthreads();

    // conv3 + residual + relu: n[o,mm] = relu(v[o,mm] + b3[o] + sum_c w3[o,c] v[c,mm])
    float* nb = nbuf + (long)b * NV;
    for (int idx = tid; idx < NV; idx += 1024) {
        int o  = idx >> 8;          // /C2
        int mm = idx & (C2 - 1);
        float acc = vb[idx] + b3[o];
        for (int c = 0; c < C4; ++c) acc += w3[o * C4 + c] * vb[c * C2 + mm];
        nb[idx] = fmaxf(acc, 0.f);
    }
    __threadfence(); __syncthreads();

    // conv4 on transposed n: z[o2,k] = b4[o2] + sum_m w4[o2,m] n[k,m] -> bf16 [C2][C4]
    __bf16* zb = z + (long)b * C2 * C4;
    for (int idx = tid; idx < NV; idx += 1024) {
        int o2 = idx >> 7;          // /C4
        int k  = idx & (C4 - 1);
        float acc = b4[o2];
        for (int mm = 0; mm < C2; ++mm) acc += w4[o2 * C2 + mm] * nb[k * C2 + mm];
        zb[idx] = f2bf(acc);
    }
}

// y = z @ x1 : yT[n][m] = sum_k z[m][k] x1T[n][k]  (K = 128)
__global__ void k_gemm_y(const __bf16* __restrict__ z, const __bf16* __restrict__ x1T,
                         __bf16* __restrict__ yT) {
    const int BM = C2 / 64, BN = HWD / 32;
    int wid  = (blockIdx.x * blockDim.x + threadIdx.x) >> 5;
    int lane = threadIdx.x & 31;
    int b  = wid / (BM * BN);
    int r  = wid % (BM * BN);
    int m0 = (r / BN) * 64, n0 = (r % BN) * 32;
    const __bf16* A  = z   + (long)b * C2 * C4;
    const __bf16* Bt = x1T + (long)b * HWD * C4;
    v8f acc[4][2];
    gemm_block<4, 2>(A, m0, C4, Bt, n0, C4, C4, lane, acc);
    __bf16* yTb = yT + (long)b * HWD * C2;
#pragma unroll
    for (int i = 0; i < 4; ++i)
#pragma unroll
        for (int j = 0; j < 2; ++j) {
            int n     = n0 + 16 * j + (lane & 15);
            int mbase = m0 + 16 * i + ((lane >> 4) << 3);
            v8bf pack;
#pragma unroll
            for (int e = 0; e < 8; ++e) pack[e] = f2bf(acc[i][j][e]);
            *(v8bf*)(yTb + (long)n * C2 + mbase) = pack;
        }
}

// conv5 + residual: out[o][n] = x[o][n] + b5[o] + sum_m w5[o][m] yT[n][m]
__global__ void k_gemm_out(const __bf16* __restrict__ w5b, const float* __restrict__ b5,
                           const __bf16* __restrict__ yT, const float* __restrict__ x,
                           float* __restrict__ out) {
    const int BM = CDIM / 64, BN = HWD / 32;
    int wid  = (blockIdx.x * blockDim.x + threadIdx.x) >> 5;
    int lane = threadIdx.x & 31;
    int b  = wid / (BM * BN);
    int r  = wid % (BM * BN);
    int m0 = (r / BN) * 64, n0 = (r % BN) * 32;
    const __bf16* Bt = yT + (long)b * HWD * C2;
    v8f acc[4][2];
    gemm_block<4, 2>(w5b, m0, C2, Bt, n0, C2, C2, lane, acc);
    const float* xb = x   + (long)b * CDIM * HWD;
    float*      ob  = out + (long)b * CDIM * HWD;
#pragma unroll
    for (int i = 0; i < 4; ++i)
#pragma unroll
        for (int j = 0; j < 2; ++j) {
            int n     = n0 + 16 * j + (lane & 15);
            int mbase = m0 + 16 * i + ((lane >> 4) << 3);
#pragma unroll
            for (int e = 0; e < 8; ++e) {
                long idx = (long)(mbase + e) * HWD + n;
                ob[idx] = xb[idx] + acc[i][j][e] + b5[mbase + e];
            }
        }
}

// ---------------- host launcher ----------------
extern "C" void kernel_launch(void* const* d_in, const int* in_sizes, int n_in,
                              void* d_out, int out_size, void* d_ws, size_t ws_size,
                              hipStream_t stream) {
    const float* x  = (const float*)d_in[0];
    const float* w1 = (const float*)d_in[1];
    const float* b1 = (const float*)d_in[2];
    const float* w2 = (const float*)d_in[3];
    const float* b2 = (const float*)d_in[4];
    const float* w3 = (const float*)d_in[5];
    const float* b3 = (const float*)d_in[6];
    const float* w4 = (const float*)d_in[7];
    const float* b4 = (const float*)d_in[8];
    const float* w5 = (const float*)d_in[9];
    const float* b5 = (const float*)d_in[10];
    float* out = (float*)d_out;

    // workspace carve-up (all sizes multiples of 256 bytes)
    char* ws = (char*)d_ws;
    size_t o = 0;
    auto carve = [&](size_t bytes) { char* p = ws + o; o += bytes; return p; };
    __bf16* xT   = (__bf16*)carve((size_t)BATCH * HWD * CDIM * 2);  // 75.5 MB
    __bf16* x1   = (__bf16*)carve((size_t)BATCH * C4  * HWD  * 2);  // 18.9 MB
    __bf16* x1T  = (__bf16*)carve((size_t)BATCH * HWD * C4   * 2);  // 18.9 MB
    __bf16* x2   = (__bf16*)carve((size_t)BATCH * C2  * HWD  * 2);  // 37.7 MB
    __bf16* yT   = (__bf16*)carve((size_t)BATCH * HWD * C2   * 2);  // 37.7 MB
    float*  v    = (float*) carve((size_t)BATCH * C4  * C2   * 4);  // 1 MB
    float*  nbuf = (float*) carve((size_t)BATCH * C4  * C2   * 4);  // 1 MB
    __bf16* z    = (__bf16*)carve((size_t)BATCH * C2  * C4   * 2);  // 0.5 MB
    __bf16* w1b  = (__bf16*)carve((size_t)C4 * CDIM * 2);
    __bf16* w2b  = (__bf16*)carve((size_t)C2 * CDIM * 2);
    __bf16* w5b  = (__bf16*)carve((size_t)CDIM * C2 * 2);
    (void)ws_size; (void)n_in; (void)in_sizes; (void)out_size;

    // 1) weight conversion to bf16
    k_cvt<<<(C4 * CDIM + 255) / 256, 256, 0, stream>>>(w1, w1b, C4 * CDIM);
    k_cvt<<<(C2 * CDIM + 255) / 256, 256, 0, stream>>>(w2, w2b, C2 * CDIM);
    k_cvt<<<(CDIM * C2 + 255) / 256, 256, 0, stream>>>(w5, w5b, CDIM * C2);

    // 2) x -> xT (bf16, [B][HW][C])
    k_transpose<<<dim3(HWD / 32, CDIM / 32, BATCH), dim3(32, 32), 0, stream>>>(x, xT);

    // 3) conv1 / conv2 (WMMA, 64x32 per wave)
    k_gemm_x1<<<BATCH * (C4 / 64) * (HWD / 32) / 8, 256, 0, stream>>>(w1b, b1, xT, x1, x1T);
    k_gemm_x2<<<BATCH * (C2 / 64) * (HWD / 32) / 8, 256, 0, stream>>>(w2b, b2, xT, x2);

    // 4) Gram matrix v = x1 x2^T (WMMA, K=9216, 1 wave per tile)
    k_gemm_v<<<BATCH * (C4 / 16) * (C2 / 16) / 8, 256, 0, stream>>>(x1, x2, v);

    // 5) softmax + conv3(+res,relu) + conv4 -> z (bf16)
    k_small<<<BATCH, 1024, 0, stream>>>(v, w3, b3, w4, b4, nbuf, z);

    // 6) y = z @ x1 (WMMA, K=128), stored transposed
    k_gemm_y<<<BATCH * (C2 / 64) * (HWD / 32) / 8, 256, 0, stream>>>(z, x1T, yT);

    // 7) conv5 + residual -> out (WMMA, K=256)
    k_gemm_out<<<BATCH * (CDIM / 64) * (HWD / 32) / 8, 256, 0, stream>>>(w5b, b5, yT, x, out);
}
